// TransformerLayer_28716151341183
// MI455X (gfx1250) — compile-verified, matrix operand order checked by reading, not run on hardware
//
#include <hip/hip_runtime.h>
#include <hip/hip_bf16.h>

#define QLEN 1024
#define MLEN 1024
#define KLEN 2048
#define PLEN 3072
#define BSZ 2
#define NH 12
#define DH 64
#define DM 768
#define DFF 3072

typedef __attribute__((ext_vector_type(16))) unsigned short u16x16;
typedef __attribute__((ext_vector_type(16))) __bf16         bf16x16;
typedef __attribute__((ext_vector_type(8)))  float          f32x8;

#if defined(__has_builtin)
#if __has_builtin(__builtin_amdgcn_global_load_async_to_lds_b128) && \
    __has_builtin(__builtin_amdgcn_s_wait_asynccnt)
#define USE_ASYNC 1
#endif
#endif

typedef int v4i_ __attribute__((vector_size(16)));
typedef __attribute__((address_space(1))) v4i_ gv4i;
typedef __attribute__((address_space(3))) v4i_ lv4i;

__device__ __forceinline__ unsigned short f2bf(float f) {
    union { float f; unsigned int u; } v; v.f = f;
    unsigned int u = v.u;
    unsigned int r = (u + 0x7FFFu + ((u >> 16) & 1u)) >> 16;
    return (unsigned short)r;
}

__device__ __forceinline__ f32x8 wmma_bf16(u16x16 a, u16x16 b, f32x8 c) {
    return __builtin_amdgcn_wmma_f32_16x16x32_bf16(
        false, __builtin_bit_cast(bf16x16, a),
        false, __builtin_bit_cast(bf16x16, b),
        (short)0, c, false, false);
}

// 16-byte global -> LDS copy (async on CDNA5 if available)
__device__ __forceinline__ void cp16(void* lds, const void* g) {
#ifdef USE_ASYNC
    __builtin_amdgcn_global_load_async_to_lds_b128((gv4i*)g, (lv4i*)lds, 0, 0);
#else
    struct U4 { unsigned int x, y, z, w; };
    *(U4*)lds = *(const U4*)g;
#endif
}
__device__ __forceinline__ void async_wait0() {
#ifdef USE_ASYNC
    __builtin_amdgcn_s_wait_asynccnt(0);
#endif
}
__device__ __forceinline__ void async_wait_prev() {   // <= one tile's copies in flight
#ifdef USE_ASYNC
    __builtin_amdgcn_s_wait_asynccnt(4);
#endif
}

// ---------------------------------------------------------------------------
// fp32 -> bf16 conversion kernels (run once)
// ---------------------------------------------------------------------------
__global__ __launch_bounds__(256) void cvt_bf16_kernel(
    const float* __restrict__ src, unsigned short* __restrict__ dst, size_t n) {
    size_t i = (size_t)blockIdx.x * 256 + threadIdx.x;
    if (i < n) dst[i] = f2bf(src[i]);
}
// dst[c*R + r] = bf16(src[r*C + c])   (src [R][C] -> dst [C][R])
__global__ __launch_bounds__(256) void cvtT_bf16_kernel(
    const float* __restrict__ src, unsigned short* __restrict__ dst, int R, int C) {
    size_t i = (size_t)blockIdx.x * 256 + threadIdx.x;
    if (i < (size_t)R * C) {
        int r = (int)(i / C), c = (int)(i % C);
        dst[(size_t)c * R + r] = f2bf(src[i]);
    }
}

// ---------------------------------------------------------------------------
// bf16 WMMA GEMM. A: [M,K] bf16 (lda=K-stride). Bt: [N,K] bf16 (ldb=K-stride).
// Outputs: C fp32 (opt), Cbf bf16 copy (opt), CbfT bf16 per-batch transposed
//   copy (opt): CbfT[((row&1)*DM + col)*ldT + (row>>1) + trow0]
// act: 0 none, 1 +bias, 2 +bias & exact GELU.  M%128==0, N%128==0, K%32==0.
// Both LDS tiles are [rows][K] so ALL fragments load as 2x ds_load_b128.
// ---------------------------------------------------------------------------
__global__ __launch_bounds__(256) void gemm_bf16_kernel(
    const unsigned short* __restrict__ A, int lda,
    const unsigned short* __restrict__ Bt, int ldb,
    float* __restrict__ C, unsigned short* __restrict__ Cbf,
    unsigned short* __restrict__ CbfT, int ldT, int trow0,
    int ldc, int M, int N, int K,
    const float* __restrict__ bias, int act)
{
    __shared__ unsigned short As[2][128][32];   // [m][k]
    __shared__ unsigned short Bs[2][128][32];   // [n][k]

    const int m0 = blockIdx.y * 128;
    const int n0 = blockIdx.x * 128;
    const int tid = threadIdx.x;
    const int w  = tid >> 5;
    const int l  = tid & 31;
    const int lh = l >> 4;
    const int ll = l & 15;
    const int wm = w & 3;
    const int wn = w >> 2;

    f32x8 acc[2][4] = {};

    auto issue = [&](int buf, int k0) {
        // A tile: 128 rows x 4 chunks (16B)
        for (int x = tid; x < 512; x += 256) {
            int row = x >> 2, cb = (x & 3) * 8;
            cp16(&As[buf][row][cb], A + (size_t)(m0 + row) * lda + k0 + cb);
        }
        // B tile: 128 rows x 4 chunks (16B), row-major [n][k]
        for (int x = tid; x < 512; x += 256) {
            int row = x >> 2, cb = (x & 3) * 8;
            cp16(&Bs[buf][row][cb], Bt + (size_t)(n0 + row) * ldb + k0 + cb);
        }
    };

    const int nk = K / 32;
    issue(0, 0);
    for (int t = 0; t < nk; ++t) {
        if (t + 1 < nk) {
            issue((t + 1) & 1, (t + 1) * 32);
            __builtin_prefetch(A + (size_t)(m0 + (tid >> 1)) * lda + (t + 1) * 32, 0, 1);
            async_wait_prev();
        } else {
            async_wait0();
        }
        __syncthreads();
        const int buf = t & 1;
        const int kb = lh * 8;
        u16x16 af[2], bfv[4];
#pragma unroll
        for (int mt = 0; mt < 2; ++mt) {
            int row = wm * 32 + mt * 16 + ll;
#pragma unroll
            for (int e = 0; e < 16; ++e) {
                int kk = kb + ((e < 8) ? e : 8 + e);
                af[mt][e] = As[buf][row][kk];
            }
        }
#pragma unroll
        for (int nt = 0; nt < 4; ++nt) {
            int col = wn * 64 + nt * 16 + ll;
#pragma unroll
            for (int e = 0; e < 16; ++e) {
                int kk = kb + ((e < 8) ? e : 8 + e);
                bfv[nt][e] = Bs[buf][col][kk];
            }
        }
#pragma unroll
        for (int mt = 0; mt < 2; ++mt)
#pragma unroll
            for (int nt = 0; nt < 4; ++nt)
                acc[mt][nt] = wmma_bf16(af[mt], bfv[nt], acc[mt][nt]);
        __syncthreads();
    }

#pragma unroll
    for (int mt = 0; mt < 2; ++mt)
#pragma unroll
        for (int nt = 0; nt < 4; ++nt)
#pragma unroll
            for (int k = 0; k < 8; ++k) {
                int row = m0 + wm * 32 + mt * 16 + k + lh * 8;
                int col = n0 + wn * 64 + nt * 16 + ll;
                float v = acc[mt][nt][k];
                if (act >= 1) v += bias[col];
                if (act == 2) v = 0.5f * v * (1.0f + erff(v * 0.70710678118654752f));
                if (C)   C[(size_t)row * ldc + col] = v;
                if (Cbf) Cbf[(size_t)row * ldc + col] = f2bf(v);
                if (CbfT)
                    CbfT[((size_t)((row & 1) * DM + col)) * ldT + (row >> 1) + trow0] = f2bf(v);
            }
}

// ---------------------------------------------------------------------------
// Flash-style XLNet relative attention, bf16 operands, async LDS tile copies.
// khb/krb: normal bf16 head tensors [jb][c]; vhT: [b][c][j] transposed bf16.
// LDS tiles are all fragment-major: Kst[j][d], KRst[r][d], Vst[d][j].
// grid = (QLEN/64, BSZ*NH), block = 128 (4 waves x 16 query rows).
// ---------------------------------------------------------------------------
__global__ __launch_bounds__(128) void attn_kernel(
    const float* __restrict__ qh, const unsigned short* __restrict__ khb,
    const unsigned short* __restrict__ vhT, const unsigned short* __restrict__ krb,
    const float* __restrict__ rwb, const float* __restrict__ rrb,
    const float* __restrict__ rsb, const float* __restrict__ segemb,
    const float* __restrict__ segmat, unsigned short* __restrict__ avb)
{
    __shared__ unsigned short Kst[32][DH];   // [j][d]
    __shared__ unsigned short KRst[96][DH];  // [rwin][d]
    __shared__ unsigned short Vst[DH][32];   // [d][j]
    __shared__ unsigned short Ps[4][16][32];
    __shared__ float Ssc[4][16][33];
    __shared__ float rowscale[4][16];
    __shared__ float rowl[4][16];
    __shared__ float efs[64][2];

    const int qbase = blockIdx.x * 64;
    const int head  = blockIdx.y % NH;
    const int b     = blockIdx.y / NH;
    const int tid = threadIdx.x;
    const int w  = tid >> 5;
    const int l  = tid & 31;
    const int lh = l >> 4;
    const int ll = l & 15;
    const int i0w = qbase + w * 16;

    // EF term per query row
    if (tid < 64) {
        int i = qbase + tid;
        const float* qrow = qh + ((size_t)(i * BSZ + b)) * DM + head * DH;
        float e0 = 0.f, e1 = 0.f;
#pragma unroll 8
        for (int d = 0; d < DH; ++d) {
            float qv = qrow[d] + rsb[head * DH + d];
            e0 += qv * segemb[(0 * NH + head) * DH + d];
            e1 += qv * segemb[(1 * NH + head) * DH + d];
        }
        efs[tid][0] = e0; efs[tid][1] = e1;
    }

    // q A-fragments (q + r_w_bias), (q + r_r_bias)
    u16x16 qc[2], qr[2];
    {
        int row = i0w + ll;
        const float* qrow = qh + ((size_t)(row * BSZ + b)) * DM + head * DH;
        int kb = lh * 8;
#pragma unroll
        for (int c = 0; c < 2; ++c)
#pragma unroll
            for (int e = 0; e < 16; ++e) {
                int d = c * 32 + kb + ((e < 8) ? e : 8 + e);
                float qv = qrow[d];
                qc[c][e] = f2bf(qv + rwb[head * DH + d]);
                qr[c][e] = f2bf(qv + rrb[head * DH + d]);
            }
    }

    f32x8 o[4] = {};
    float mrow = -INFINITY, lrowv = 0.f;

    const unsigned short* vRow = vhT + (size_t)(b * DM + head * DH) * KLEN;

    const int ntiles = (qbase + 64 + MLEN) / 32;
    for (int t = 0; t < ntiles; ++t) {
        const int j0 = t * 32;
        const int rb0 = j0 - qbase - 63 + QLEN;   // kr window base, in [1, PLEN-96]
        __syncthreads();
        // K tile: 32 rows x 8 chunks
        for (int x = tid; x < 256; x += 128) {
            int j = x >> 3, cb = (x & 7) * 8;
            cp16(&Kst[j][cb], khb + ((size_t)((j0 + j) * BSZ + b)) * DM + head * DH + cb);
        }
        // KR window: 96 rows x 8 chunks
        for (int x = tid; x < 768; x += 128) {
            int rw = x >> 3, cb = (x & 7) * 8;
            cp16(&KRst[rw][cb], krb + ((size_t)((rb0 + rw) * BSZ + b)) * DM + head * DH + cb);
        }
        // V tile: 64 rows x 4 chunks
        for (int x = tid; x < 256; x += 128) {
            int d = x >> 2, cb = (x & 3) * 8;
            cp16(&Vst[d][cb], vRow + (size_t)d * KLEN + j0 + cb);
        }
        async_wait0();
        __syncthreads();

        // ---- AC ----
#pragma unroll
        for (int nt2 = 0; nt2 < 2; ++nt2) {
            f32x8 acc = {};
#pragma unroll
            for (int kc = 0; kc < 2; ++kc) {
                u16x16 bf;
                int col = nt2 * 16 + ll;
                int kb = kc * 32 + lh * 8;
#pragma unroll
                for (int e = 0; e < 16; ++e) {
                    int d = kb + ((e < 8) ? e : 8 + e);
                    bf[e] = Kst[col][d];
                }
                acc = wmma_bf16(qc[kc], bf, acc);
            }
#pragma unroll
            for (int k = 0; k < 8; ++k)
                Ssc[w][k + lh * 8][nt2 * 16 + ll] = acc[k];
        }

        // ---- BD: diagonal band, scatter j = c + m - 15 ----
        const int cbase = 48 - 16 * w;   // rwin base for this wave
#pragma unroll
        for (int ct = 0; ct < 3; ++ct) {
            f32x8 acc = {};
#pragma unroll
            for (int kc = 0; kc < 2; ++kc) {
                u16x16 bf;
                int rw = cbase + ct * 16 + ll;
                int kb = kc * 32 + lh * 8;
#pragma unroll
                for (int e = 0; e < 16; ++e) {
                    int d = kb + ((e < 8) ? e : 8 + e);
                    bf[e] = KRst[rw][d];
                }
                acc = wmma_bf16(qr[kc], bf, acc);
            }
#pragma unroll
            for (int k = 0; k < 8; ++k) {
                int m  = k + lh * 8;
                int c  = ct * 16 + ll;
                int jl = c + m - 15;
                if (jl >= 0 && jl < 32) Ssc[w][m][jl] += acc[k];
            }
        }

        // ---- EF + scale + mask + online softmax ----
        {
            float e0 = efs[w * 16 + ll][0];
            float e1 = efs[w * 16 + ll][1];
            int i = i0w + ll;
            float cmax = -INFINITY;
            float sv[16];
#pragma unroll
            for (int c = 0; c < 16; ++c) {
                int jl = lh * 16 + c;
                int j  = j0 + jl;
                float s = Ssc[w][ll][jl];
                size_t sb = (((size_t)i * KLEN + j) * BSZ + b) * 2;
                s += segmat[sb] * e0 + segmat[sb + 1] * e1;
                s *= 0.125f;
                if (j > i + MLEN) s = -1e30f;
                sv[c] = s;
                cmax = fmaxf(cmax, s);
            }
            float omax = fmaxf(cmax, __shfl_xor(cmax, 16, 32));
            float mnew = fmaxf(mrow, omax);
            float csum = 0.f;
#pragma unroll
            for (int c = 0; c < 16; ++c) {
                float p = expf(sv[c] - mnew);
                csum += p;
                Ps[w][ll][lh * 16 + c] = f2bf(p);
            }
            float osum = csum + __shfl_xor(csum, 16, 32);
            float sc = expf(mrow - mnew);
            lrowv = lrowv * sc + osum;
            mrow = mnew;
            if (lh == 0) rowscale[w][ll] = sc;
        }

        // ---- rescale O, accumulate P.V ----
#pragma unroll
        for (int nt = 0; nt < 4; ++nt)
#pragma unroll
            for (int k = 0; k < 8; ++k)
                o[nt][k] *= rowscale[w][k + lh * 8];
        {
            u16x16 pa;
            int kb = lh * 8;
#pragma unroll
            for (int e = 0; e < 16; ++e) {
                int kk = kb + ((e < 8) ? e : 8 + e);
                pa[e] = Ps[w][ll][kk];
            }
#pragma unroll
            for (int nt = 0; nt < 4; ++nt) {
                u16x16 bf;
                int dcol = nt * 16 + ll;
#pragma unroll
                for (int e = 0; e < 16; ++e) {
                    int kk = kb + ((e < 8) ? e : 8 + e);
                    bf[e] = Vst[dcol][kk];
                }
                o[nt] = wmma_bf16(pa, bf, o[nt]);
            }
        }
    }

    if (lh == 0) rowl[w][ll] = lrowv;
#pragma unroll
    for (int nt = 0; nt < 4; ++nt)
#pragma unroll
        for (int k = 0; k < 8; ++k) {
            int m = k + lh * 8;
            int d = nt * 16 + ll;
            int i = i0w + m;
            avb[((size_t)(i * BSZ + b)) * DM + head * DH + d] = f2bf(o[nt][k] / rowl[w][m]);
        }
}

// ---------------------------------------------------------------------------
// out = LayerNorm(x + res); optional bf16 copy for downstream GEMM A operand.
// ---------------------------------------------------------------------------
__global__ __launch_bounds__(256) void ln_kernel(
    const float* __restrict__ x, const float* __restrict__ res,
    const float* __restrict__ g, const float* __restrict__ bb,
    float* __restrict__ out, unsigned short* __restrict__ out_bf, int D)
{
    __shared__ float red[256];
    const int row = blockIdx.x;
    const int tid = threadIdx.x;
    const float* xp = x   + (size_t)row * D;
    const float* rp = res + (size_t)row * D;
    float s = 0.f;
    for (int d = tid; d < D; d += 256) s += xp[d] + rp[d];
    red[tid] = s; __syncthreads();
    for (int st = 128; st > 0; st >>= 1) { if (tid < st) red[tid] += red[tid + st]; __syncthreads(); }
    float mu = red[0] / D; __syncthreads();
    float v = 0.f;
    for (int d = tid; d < D; d += 256) { float t = xp[d] + rp[d] - mu; v += t * t; }
    red[tid] = v; __syncthreads();
    for (int st = 128; st > 0; st >>= 1) { if (tid < st) red[tid] += red[tid + st]; __syncthreads(); }
    float rstd = rsqrtf(red[0] / D + 1e-12f);
    for (int d = tid; d < D; d += 256) {
        float y = (xp[d] + rp[d] - mu) * rstd * g[d] + bb[d];
        if (out)    out[(size_t)row * D + d] = y;
        if (out_bf) out_bf[(size_t)row * D + d] = f2bf(y);
    }
}

// ---------------------------------------------------------------------------
extern "C" void kernel_launch(void* const* d_in, const int* in_sizes, int n_in,
                              void* d_out, int out_size, void* d_ws, size_t ws_size,
                              hipStream_t stream) {
    (void)in_sizes; (void)n_in; (void)out_size; (void)ws_size;
    const float* h      = (const float*)d_in[0];
    const float* mem    = (const float*)d_in[1];
    const float* pos    = (const float*)d_in[2];
    const float* segmat = (const float*)d_in[3];
    const float* q_w  = (const float*)d_in[5];
    const float* k_w  = (const float*)d_in[6];
    const float* v_w  = (const float*)d_in[7];
    const float* o_w  = (const float*)d_in[8];
    const float* r_w  = (const float*)d_in[9];
    const float* rwb  = (const float*)d_in[10];
    const float* rrb  = (const float*)d_in[11];
    const float* rsb  = (const float*)d_in[12];
    const float* sege = (const float*)d_in[13];
    const float* ln1g = (const float*)d_in[14];
    const float* ln1b = (const float*)d_in[15];
    const float* ffw1 = (const float*)d_in[16];
    const float* ffb1 = (const float*)d_in[17];
    const float* ffw2 = (const float*)d_in[18];
    const float* ffb2 = (const float*)d_in[19];
    const float* ln2g = (const float*)d_in[20];
    const float* ln2b = (const float*)d_in[21];
    float* out = (float*)d_out;

    const int MQ = QLEN * BSZ;     // 2048
    const int MK = KLEN * BSZ;     // 4096
    const int MP = PLEN * BSZ;     // 6144

    char* p = (char*)d_ws;
    auto allocF = [&](size_t n) { float* r = (float*)p; p += ((n * 4 + 255) & ~(size_t)255); return r; };
    auto allocH = [&](size_t n) { unsigned short* r = (unsigned short*)p; p += ((n * 2 + 255) & ~(size_t)255); return r; };

    float* qh  = allocF((size_t)MQ * DM);
    float* ao  = allocF((size_t)MQ * DM);
    float* oh  = allocF((size_t)MQ * DM);
    float* ffo = allocF((size_t)MQ * DM);

    unsigned short* hb   = allocH((size_t)MQ * DM);   // activations, [row][k]
    unsigned short* memb = allocH((size_t)MQ * DM);
    unsigned short* posb = allocH((size_t)MP * DM);
    unsigned short* qwT  = allocH((size_t)DM * DM);   // weights, [n][k]
    unsigned short* kwT  = allocH((size_t)DM * DM);
    unsigned short* vwT  = allocH((size_t)DM * DM);
    unsigned short* rwT  = allocH((size_t)DM * DM);
    unsigned short* owN  = allocH((size_t)DM * DM);   // o_w already [h][k] == [n][k]
    unsigned short* w1T  = allocH((size_t)DFF * DM);  // [n=DFF][k=DM]
    unsigned short* w2T  = allocH((size_t)DM * DFF);  // [n=DM][k=DFF]
    unsigned short* khb  = allocH((size_t)MK * DM);   // k heads, normal [jb][c]
    unsigned short* krb  = allocH((size_t)MP * DM);   // r heads, normal [pb][c]
    unsigned short* vhT  = allocH((size_t)BSZ * DM * KLEN);  // v heads transposed [b][c][j]
    unsigned short* avbp = allocH((size_t)MQ * DM);
    unsigned short* ohb  = allocH((size_t)MQ * DM);
    unsigned short* ffab = allocH((size_t)MQ * DFF);

    // one-time bf16 conversions
    cvt_bf16_kernel<<<(MQ*DM)/256, 256, 0, stream>>>(h,    hb,   (size_t)MQ*DM);
    cvt_bf16_kernel<<<(MQ*DM)/256, 256, 0, stream>>>(mem,  memb, (size_t)MQ*DM);
    cvt_bf16_kernel<<<(MP*DM)/256, 256, 0, stream>>>(pos,  posb, (size_t)MP*DM);
    cvtT_bf16_kernel<<<(DM*DM)/256, 256, 0, stream>>>(q_w,  qwT, DM, DM);
    cvtT_bf16_kernel<<<(DM*DM)/256, 256, 0, stream>>>(k_w,  kwT, DM, DM);
    cvtT_bf16_kernel<<<(DM*DM)/256, 256, 0, stream>>>(v_w,  vwT, DM, DM);
    cvtT_bf16_kernel<<<(DM*DM)/256, 256, 0, stream>>>(r_w,  rwT, DM, DM);
    cvt_bf16_kernel<<<(DM*DM)/256, 256, 0, stream>>>(o_w,  owN, (size_t)DM*DM);
    cvtT_bf16_kernel<<<(DM*DFF)/256, 256, 0, stream>>>(ffw1, w1T, DM, DFF);
    cvtT_bf16_kernel<<<(DFF*DM)/256, 256, 0, stream>>>(ffw2, w2T, DFF, DM);

    // projections
    gemm_bf16_kernel<<<dim3(DM/128, MQ/128), 256, 0, stream>>>(hb,   DM, qwT, DM, qh, nullptr, nullptr, 1, 0, DM, MQ, DM, DM, nullptr, 0);
    gemm_bf16_kernel<<<dim3(DM/128, MQ/128), 256, 0, stream>>>(memb, DM, kwT, DM, nullptr, khb,                 nullptr, 1, 0, DM, MQ, DM, DM, nullptr, 0);
    gemm_bf16_kernel<<<dim3(DM/128, MQ/128), 256, 0, stream>>>(hb,   DM, kwT, DM, nullptr, khb + (size_t)MQ*DM, nullptr, 1, 0, DM, MQ, DM, DM, nullptr, 0);
    gemm_bf16_kernel<<<dim3(DM/128, MQ/128), 256, 0, stream>>>(memb, DM, vwT, DM, nullptr, nullptr, vhT, KLEN, 0,    DM, MQ, DM, DM, nullptr, 0);
    gemm_bf16_kernel<<<dim3(DM/128, MQ/128), 256, 0, stream>>>(hb,   DM, vwT, DM, nullptr, nullptr, vhT, KLEN, 1024, DM, MQ, DM, DM, nullptr, 0);
    gemm_bf16_kernel<<<dim3(DM/128, MP/128), 256, 0, stream>>>(posb, DM, rwT, DM, nullptr, krb,                 nullptr, 1, 0, DM, MP, DM, DM, nullptr, 0);

    // relative attention
    attn_kernel<<<dim3(QLEN/64, BSZ*NH), 128, 0, stream>>>(qh, khb, vhT, krb, rwb, rrb, rsb, sege, segmat, avbp);

    // output projection
    gemm_bf16_kernel<<<dim3(DM/128, MQ/128), 256, 0, stream>>>(avbp, DM, owN, DM, ao, nullptr, nullptr, 1, 0, DM, MQ, DM, DM, nullptr, 0);

    // residual + LN1 (fp32 out + bf16 copy)
    ln_kernel<<<MQ, 256, 0, stream>>>(ao, h, ln1g, ln1b, oh, ohb, DM);

    // FF
    gemm_bf16_kernel<<<dim3(DFF/128, MQ/128), 256, 0, stream>>>(ohb,  DM,  w1T, DM,  nullptr, ffab, nullptr, 1, 0, DFF, MQ, DFF, DM,  ffb1, 2);
    gemm_bf16_kernel<<<dim3(DM/128,  MQ/128), 256, 0, stream>>>(ffab, DFF, w2T, DFF, ffo,     nullptr, nullptr, 1, 0, DM, MQ, DM,  DFF, ffb2, 1);

    // residual + LN2 -> output
    ln_kernel<<<MQ, 256, 0, stream>>>(ffo, oh, ln2g, ln2b, out, nullptr, DM);
}
